// RelationExtractor_55473797595941
// MI455X (gfx1250) — matrix operand order repaired and checked by reading, not compiled, not run on hardware
//
#include <hip/hip_runtime.h>
#include <hip/hip_bf16.h>

// ---------------------------------------------------------------------------
// RelationExtractor for MI455X (gfx1250, wave32, WMMA + async Global->LDS).
//   N=512 entities, D=768, H=256, T=16, B=4, THRESH=0.5, MAXREL=5
// Pipeline:
//   1) convert e and concatenated W1 halves to f16
//   2) WMMA GEMM: Hmat[512,1536] = e[512,768] @ Wcat[768,1536]  (f16 in, f32 acc)
//   3) pair kernel: stage 16 j-rows (48KB) via global_load_async_to_lds_b128,
//      s_wait_asynccnt; LN/GELU/score/dir via VALU + wave32 reductions;
//      classifier logits via v_wmma_f32_16x16x32_f16 (LDS-staged f16 A-tiles)
//      also emits sort keys (~score_bits, idx) and zeroes `selected`
//   4) bitonic sort of 2^18 u64 keys (fixed 171 launches)
//   5) single-wave serial greedy (counts<5 per entity) -> selected=1.0
// ---------------------------------------------------------------------------

#define NN_ENT 512
#define DD 768
#define HH 256
#define TT 16
#define WROW 1536          // Hmat row width (3 heads x 2 halves x 256)
#define NPAIR (NN_ENT * NN_ENT)

// dynamic LDS partition for pair_kernel (gfx1250: 320KB/WGP, so 88KB is fine)
#define LDS_HJ_BYTES   49152                    // 3 chunks x 16 rows x 256 f32
#define LDS_B_BYTES    8192                     // cW2^T f16 [16][256]
#define LDS_A_BYTES    32768                    // 4 waves x [16][256] f16
#define LDS_TOTAL      (LDS_HJ_BYTES + LDS_B_BYTES + LDS_A_BYTES)

typedef __attribute__((ext_vector_type(16))) _Float16 v16h;
typedef __attribute__((ext_vector_type(8)))  float    v8f;

union F16x16 { v16h v; uint4 u4[2]; };

__device__ __forceinline__ float wredsum(float v) {
#pragma unroll
  for (int off = 16; off >= 1; off >>= 1) v += __shfl_xor(v, off, 32);
  return v;
}

__device__ __forceinline__ float gelu_exact(float x) {
  return 0.5f * x * (1.0f + erff(x * 0.70710678118654752f));
}

__device__ __forceinline__ float sigmoidf_(float x) {
  return 1.0f / (1.0f + expf(-x));
}

// async copy of 16 bytes Global -> LDS (ASYNCcnt-tracked, per-lane addresses)
__device__ __forceinline__ void async_g2l_b128(unsigned lds_off, const void* gsrc) {
  asm volatile("global_load_async_to_lds_b128 %0, %1, off"
               :: "v"(lds_off), "v"((unsigned long long)(uintptr_t)gsrc)
               : "memory");
}

__device__ __forceinline__ void wait_asynccnt0() {
  asm volatile("s_wait_asynccnt 0x0" ::: "memory");
}

// ---------------------------------------------------------------------------
// 1) fp32 -> f16 conversion of entity matrix
__global__ void conv_e_kernel(const float* __restrict__ e, _Float16* __restrict__ e16) {
  int idx = blockIdx.x * blockDim.x + threadIdx.x;
  if (idx < NN_ENT * DD) e16[idx] = (_Float16)e[idx];
}

// Build Wcat transposed (n-major, k-contiguous): w16T[n*768 + k]
// column n: head t = n/512, half hf = (n/256)&1, col = n%256
// source W1_t shape (1536,256) row-major -> W1_t[(hf*768+k)*256 + col]
__global__ void conv_w_kernel(const float* __restrict__ sW1,
                              const float* __restrict__ cW1,
                              const float* __restrict__ dW1,
                              _Float16* __restrict__ w16T) {
  int idx = blockIdx.x * blockDim.x + threadIdx.x;
  if (idx >= WROW * DD) return;
  int n = idx / DD;
  int k = idx - n * DD;
  int t = n >> 9;
  int hf = (n >> 8) & 1;
  int col = n & 255;
  const float* W = (t == 0) ? sW1 : (t == 1) ? cW1 : dW1;
  w16T[idx] = (_Float16)W[(hf * DD + k) * HH + col];
}

// ---------------------------------------------------------------------------
// 2) WMMA GEMM: Hmat = e16[512,768] @ Wcat[768,1536], one wave per 16x16 tile.
__global__ __launch_bounds__(32) void gemm_kernel(const _Float16* __restrict__ A16,  // [512][768]
                                                  const _Float16* __restrict__ BT,   // [1536][768] n-major
                                                  float* __restrict__ C) {           // [512][1536]
  const int m0 = blockIdx.x * 16;
  const int n0 = blockIdx.y * 16;
  const int lane = threadIdx.x;
  const int nm  = lane & 15;
  const int kbA = (lane & 16) ? 8 : 0;    // A 16x32 f16 layout: hi-lanes hold K+8
  const int kbB = (lane & 16) ? 16 : 0;   // unused; B loads use kbA (hi-lanes hold K+8)

  const _Float16* arow = A16 + (m0 + nm) * DD;
  const _Float16* brow = BT + (n0 + nm) * DD;
  v8f acc = {0.f, 0.f, 0.f, 0.f, 0.f, 0.f, 0.f, 0.f};

#pragma unroll 4
  for (int k0 = 0; k0 < DD; k0 += 32) {
    F16x16 af, bf;
    af.u4[0] = *(const uint4*)(arow + k0 + kbA);        // K = k0+kb .. +7
    af.u4[1] = *(const uint4*)(arow + k0 + 16 + kbA);   // K = k0+16+kb .. +7
    bf.u4[0] = *(const uint4*)(brow + k0 + kbA);        // K = k0+kb .. +7
    bf.u4[1] = *(const uint4*)(brow + k0 + 16 + kbA);   // K = k0+16+kb .. +7
    acc = __builtin_amdgcn_wmma_f32_16x16x32_f16(false, af.v, false, bf.v,
                                                 (short)0, acc, false, false);
  }
#pragma unroll
  for (int r = 0; r < 8; ++r) {
    int m = m0 + r + ((lane & 16) ? 8 : 0);             // C layout: VGPR r -> M=r / r+8
    C[m * WROW + n0 + nm] = acc[r];
  }
}

// ---------------------------------------------------------------------------
// 3) Pair kernel. Block = 16x16 pair tile, 128 threads = 4 waves, 4 i-rows/wave.
//    Dynamic LDS: [hj staging 48KB fp32][cW2^T 8KB f16][4x A-staging 32KB f16]
__global__ __launch_bounds__(128) void pair_kernel(
    const float* __restrict__ Hmat, const int* __restrict__ batch,
    const float* __restrict__ sb1, const float* __restrict__ s_g,
    const float* __restrict__ s_b, const float* __restrict__ sW2,
    const float* __restrict__ sb2,
    const float* __restrict__ cb1, const float* __restrict__ c_g,
    const float* __restrict__ c_b, const float* __restrict__ cW2,
    const float* __restrict__ cb2,
    const float* __restrict__ db1, const float* __restrict__ dW2,
    const float* __restrict__ db2,
    float* __restrict__ out,              // [N][N][18]
    float* __restrict__ selected,         // [N][N]
    unsigned long long* __restrict__ keys) {

  extern __shared__ char smem[];
  float*    hjbuf = (float*)smem;                              // [3][16][256] f32
  _Float16* lds_b = (_Float16*)(smem + LDS_HJ_BYTES);          // [16][256] f16
  _Float16* lds_a = (_Float16*)(smem + LDS_HJ_BYTES + LDS_B_BYTES); // [4][16*256]

  const int tid  = threadIdx.x;
  const int wave = tid >> 5;
  const int lane = tid & 31;
  const int i0 = blockIdx.x * 16;
  const int j0 = blockIdx.y * 16;

  // ---- async-stage the 16 j-rows' hj chunks (s:+256, c:+768, d:+1280) ------
  // 3*16*256 floats = 3072 x b128; 24 async loads per thread.
  {
    const int chunkoff[3] = {256, 768, 1280};
    for (int q = tid; q < 3072; q += 128) {
      int t   = q >> 10;          // head chunk
      int rem = q & 1023;
      int jj  = rem >> 6;         // j row in tile
      int c4  = rem & 63;         // float4 index within 256-ch chunk
      const float* src = Hmat + (long long)(j0 + jj) * WROW + chunkoff[t] + c4 * 4;
      unsigned lds_off = (unsigned)(uintptr_t)(hjbuf + q * 4);
      async_g2l_b128(lds_off, src);
    }
  }

  // stage cW2 (row-major [k][n]) transposed into LDS as f16 (plain DS stores)
  for (int idx = tid; idx < HH * TT; idx += 128) {
    int k = idx >> 4, n = idx & 15;
    lds_b[n * HH + k] = (_Float16)cW2[idx];
  }

  wait_asynccnt0();      // all of this wave's async Global->LDS copies landed
  __syncthreads();       // make every wave's staging visible block-wide

  const float* hj_s = hjbuf;            // [16][256]
  const float* hj_c = hjbuf + 4096;
  const float* hj_d = hjbuf + 8192;

  // per-lane channel slice: c = lane + 32*kk, kk<8
  float g_s[8], b_s[8], bi_s[8], w2_s[8];
  float g_c[8], b_c[8], bi_c[8];
  float bi_d[8], w2_d[8];
#pragma unroll
  for (int kk = 0; kk < 8; ++kk) {
    int c = lane + 32 * kk;
    bi_s[kk] = sb1[c]; g_s[kk] = s_g[c]; b_s[kk] = s_b[c]; w2_s[kk] = sW2[c];
    bi_c[kk] = cb1[c]; g_c[kk] = c_g[c]; b_c[kk] = c_b[c];
    bi_d[kk] = db1[c]; w2_d[kk] = dW2[c];
  }
  const float sb2v = sb2[0], db2v = db2[0];
  const int nm  = lane & 15;
  const int kbA = (lane & 16) ? 8 : 0;
  const int kbB = (lane & 16) ? 16 : 0;

  for (int rr = 0; rr < 4; ++rr) {
    const int i = i0 + wave * 4 + rr;
    const float* rowi = Hmat + (long long)i * WROW;
    __builtin_prefetch(rowi + WROW, 0, 1);   // global_prefetch next i-row
    float hi_s[8], hi_c[8], hi_d[8];
#pragma unroll
    for (int kk = 0; kk < 8; ++kk) {
      int c = lane + 32 * kk;
      hi_s[kk] = rowi[c];
      hi_c[kk] = rowi[512 + c];
      hi_d[kk] = rowi[1024 + c];
    }
    const int bi = batch[i];

    for (int jj = 0; jj < 16; ++jj) {
      const int j = j0 + jj;

      float vs[8], vc[8], vd[8];
      float sum_s = 0.f, sum_c = 0.f;
#pragma unroll
      for (int kk = 0; kk < 8; ++kk) {
        int c = lane + 32 * kk;
        vs[kk] = hi_s[kk] + hj_s[jj * HH + c] + bi_s[kk];
        vc[kk] = hi_c[kk] + hj_c[jj * HH + c] + bi_c[kk];
        vd[kk] = hi_d[kk] + hj_d[jj * HH + c] + bi_d[kk];
        sum_s += vs[kk]; sum_c += vc[kk];
      }
      const float mean_s = wredsum(sum_s) * (1.f / 256.f);
      const float mean_c = wredsum(sum_c) * (1.f / 256.f);
      float var_s = 0.f, var_c = 0.f;
#pragma unroll
      for (int kk = 0; kk < 8; ++kk) {
        float ds = vs[kk] - mean_s; var_s += ds * ds;
        float dc = vc[kk] - mean_c; var_c += dc * dc;
      }
      const float rs = rsqrtf(wredsum(var_s) * (1.f / 256.f) + 1e-5f);
      const float rc = rsqrtf(wredsum(var_c) * (1.f / 256.f) + 1e-5f);

      float dot_s = 0.f, dot_d = 0.f;
#pragma unroll
      for (int kk = 0; kk < 8; ++kk) {
        float xs = gelu_exact((vs[kk] - mean_s) * rs * g_s[kk] + b_s[kk]);
        dot_s += xs * w2_s[kk];
        float xc = gelu_exact((vc[kk] - mean_c) * rc * g_c[kk] + b_c[kk]);
        lds_a[wave * 4096 + jj * HH + lane + 32 * kk] = (_Float16)xc;  // WMMA A
        float xd = gelu_exact(vd[kk]);
        dot_d += xd * w2_d[kk];
      }
      const float score = sigmoidf_(wredsum(dot_s) + sb2v);
      const float dir   = sigmoidf_(wredsum(dot_d) + db2v);

      if (lane == 0) {
        const int idx = i * NN_ENT + j;
        const long long base = (long long)idx * 18;
        out[base + 0]  = score;
        out[base + 17] = dir;
        selected[idx] = 0.0f;
        bool cand = (bi == batch[j]) && (i != j) && (score >= 0.5f);
        unsigned sbits; __builtin_memcpy(&sbits, &score, 4);
        unsigned hi32 = cand ? ~sbits : 0xFFFFFFFFu;   // desc score, sentinel last
        keys[idx] = ((unsigned long long)hi32 << 32) | (unsigned)idx;
      }
    }

    // classifier: [16 pairs x 256] @ cW2[256 x 16] via 8x WMMA 16x16x32 f16
    const _Float16* ap = lds_a + wave * 4096;
    v8f acc = {0.f, 0.f, 0.f, 0.f, 0.f, 0.f, 0.f, 0.f};
#pragma unroll
    for (int k0 = 0; k0 < HH; k0 += 32) {
      F16x16 af, bf;
      af.u4[0] = *(const uint4*)(ap + nm * HH + k0 + kbA);
      af.u4[1] = *(const uint4*)(ap + nm * HH + k0 + 16 + kbA);
      bf.u4[0] = *(const uint4*)(&lds_b[nm * HH + k0 + kbA]);
      bf.u4[1] = *(const uint4*)(&lds_b[nm * HH + k0 + 16 + kbA]);
      acc = __builtin_amdgcn_wmma_f32_16x16x32_f16(false, af.v, false, bf.v,
                                                   (short)0, acc, false, false);
    }
#pragma unroll
    for (int r = 0; r < 8; ++r) {
      int mp = r + ((lane & 16) ? 8 : 0);    // pair (j - j0)
      long long base = ((long long)i * NN_ENT + (j0 + mp)) * 18;
      out[base + 1 + nm] = acc[r] + cb2[nm];
    }
  }
}

// ---------------------------------------------------------------------------
// 4) Bitonic sort step (ascending u64 keys).
__global__ void bitonic_step(unsigned long long* __restrict__ keys, int j, int k) {
  int i = blockIdx.x * blockDim.x + threadIdx.x;
  int ixj = i ^ j;
  if (ixj > i) {
    unsigned long long a = keys[i], b = keys[ixj];
    bool up = ((i & k) == 0);
    if ((a > b) == up) { keys[i] = b; keys[ixj] = a; }
  }
}

// ---------------------------------------------------------------------------
// 5) Serial greedy (single wave, lanes prefetch 32 keys/iteration).
__global__ __launch_bounds__(32) void greedy_kernel(const unsigned long long* __restrict__ keys,
                                                    float* __restrict__ selected) {
  __shared__ int counts[NN_ENT];
  __shared__ unsigned long long buf[32];
  const int lane = threadIdx.x;
  for (int t = lane; t < NN_ENT; t += 32) counts[t] = 0;

  for (int base = 0; base < NPAIR; base += 32) {
    unsigned long long k = keys[base + lane];
    buf[lane] = k;
    bool sent = ((unsigned)(k >> 32)) == 0xFFFFFFFFu;
    if (__all(sent)) break;          // sorted: all remaining are non-candidates
    if (lane == 0) {
      for (int t = 0; t < 32; ++t) {
        unsigned long long kt = buf[t];
        if (((unsigned)(kt >> 32)) == 0xFFFFFFFFu) continue;
        int idx = (int)(kt & 0xFFFFFFFFull);
        int ii = idx >> 9, jj = idx & (NN_ENT - 1);
        if (counts[ii] < 5 && counts[jj] < 5) {
          counts[ii]++; counts[jj]++;
          selected[idx] = 1.0f;
        }
      }
    }
  }
}

// ---------------------------------------------------------------------------
extern "C" void kernel_launch(void* const* d_in, const int* in_sizes, int n_in,
                              void* d_out, int out_size, void* d_ws, size_t ws_size,
                              hipStream_t stream) {
  (void)in_sizes; (void)n_in; (void)out_size; (void)ws_size;

  const float* e     = (const float*)d_in[0];
  const int*   batch = (const int*)d_in[1];
  // d_in[2] = hidden_states (unused by reference)
  const float* sW1 = (const float*)d_in[3];
  const float* sb1 = (const float*)d_in[4];
  const float* s_g = (const float*)d_in[5];
  const float* s_b = (const float*)d_in[6];
  const float* sW2 = (const float*)d_in[7];
  const float* sb2 = (const float*)d_in[8];
  const float* cW1 = (const float*)d_in[9];
  const float* cb1 = (const float*)d_in[10];
  const float* c_g = (const float*)d_in[11];
  const float* c_b = (const float*)d_in[12];
  const float* cW2 = (const float*)d_in[13];
  const float* cb2 = (const float*)d_in[14];
  const float* dW1 = (const float*)d_in[15];
  const float* db1 = (const float*)d_in[16];
  const float* dW2 = (const float*)d_in[17];
  const float* db2 = (const float*)d_in[18];

  char* ws = (char*)d_ws;
  _Float16* e16  = (_Float16*)ws;                         // 512*768*2   = 786432
  _Float16* w16T = (_Float16*)(ws + 786432);              // 1536*768*2  = 2359296
  float*    Hmat = (float*)(ws + 3145728);                // 512*1536*4  = 3145728
  unsigned long long* keys = (unsigned long long*)(ws + 6291456); // 2^18*8 = 2097152

  float* out      = (float*)d_out;                        // [N][N][18]
  float* selected = out + (long long)NPAIR * 18;          // [N][N]

  conv_e_kernel<<<(NN_ENT * DD) / 256, 256, 0, stream>>>(e, e16);
  conv_w_kernel<<<(WROW * DD) / 256, 256, 0, stream>>>(sW1, cW1, dW1, w16T);

  gemm_kernel<<<dim3(NN_ENT / 16, WROW / 16), 32, 0, stream>>>(e16, w16T, Hmat);

  pair_kernel<<<dim3(NN_ENT / 16, NN_ENT / 16), 128, LDS_TOTAL, stream>>>(
      Hmat, batch, sb1, s_g, s_b, sW2, sb2,
      cb1, c_g, c_b, cW2, cb2, db1, dW2, db2,
      out, selected, keys);

  for (int k = 2; k <= NPAIR; k <<= 1)
    for (int j = k >> 1; j > 0; j >>= 1)
      bitonic_step<<<NPAIR / 256, 256, 0, stream>>>(keys, j, k);

  greedy_kernel<<<1, 32, 0, stream>>>(keys, selected);
}
